// RopeSelfAttention_42159398977855
// MI455X (gfx1250) — matrix operand order repaired and checked by reading, not compiled, not run on hardware
//
#include <hip/hip_runtime.h>
#include <hip/hip_bf16.h>

// Problem constants (match reference)
#define BATCH   2
#define SEQ     2048
#define DM      1024
#define NH      16
#define HD      64
#define TOKENS  (BATCH*SEQ)   // 4096
#define QKVN    (3*DM)        // 3072

typedef __attribute__((ext_vector_type(16))) __bf16 v16bf;
typedef __attribute__((ext_vector_type(8)))  __bf16 bf16x8;
typedef __attribute__((ext_vector_type(8)))  float  v8f;

// D = A(16x32 bf16) * B(32x16 bf16) + C(16x16 f32)
static __device__ __forceinline__ v8f wmma_bf16(v16bf a, v16bf b, v8f c) {
  return __builtin_amdgcn_wmma_f32_16x16x32_bf16(false, a, false, b, (short)0, c,
                                                 false, false);
}

// A-operand fragment: lane holds row M=lane&15.
// K = k0+16*half .. +15 contiguous.
static __device__ __forceinline__ v16bf load_a16(const __bf16* rowp, int k0, int half) {
  bf16x8 lo = *(const bf16x8*)(rowp + k0 + 16*half);
  bf16x8 hi = *(const bf16x8*)(rowp + k0 + 16*half + 8);
  v16bf r;
#pragma unroll
  for (int i = 0; i < 8; ++i) { r[i] = lo[i]; r[i+8] = hi[i]; }
  return r;
}

// B-operand fragment: lane holds column N=lane&15; K = k0+16*half .. +15 contiguous.
static __device__ __forceinline__ v16bf load_b16(const __bf16* colp, int k0, int half) {
  bf16x8 lo = *(const bf16x8*)(colp + k0 + 16*half);
  bf16x8 hi = *(const bf16x8*)(colp + k0 + 16*half + 8);
  v16bf r;
#pragma unroll
  for (int i = 0; i < 8; ++i) { r[i] = lo[i]; r[i+8] = hi[i]; }
  return r;
}

// ---------------- prep kernels ----------------
__global__ void k_cvt_bf16(const float* __restrict__ s, __bf16* __restrict__ d, int n) {
  int i = blockIdx.x * blockDim.x + threadIdx.x;
  if (i < n) d[i] = (__bf16)s[i];
}

// dst[n*K + k] = (bf16) src[k*N + n]   (make weight columns contiguous along K)
__global__ void k_transpose_cvt(const float* __restrict__ s, __bf16* __restrict__ d,
                                int K, int N) {
  int i = blockIdx.x * blockDim.x + threadIdx.x;
  if (i < K * N) {
    int k = i / N, n = i % N;
    d[(size_t)n * K + k] = (__bf16)s[i];
  }
}

// RoPE + scatter for one 16x16 C sub-tile of the QKV result.
// row0: global token row of sub-tile; col: this lane's global column in [0,3072).
static __device__ __forceinline__ void qkv_epilogue_tile(
    v8f acc, int row0, int col, int half,
    __bf16* __restrict__ Qb, __bf16* __restrict__ Kb, __bf16* __restrict__ Vt) {
  const int sel = col >> 10;            // 0=Q 1=K 2=V (uniform per sub-tile)
  const int cc  = col & (DM - 1);
  const int h   = cc >> 6;
  const int d   = cc & 63;
#pragma unroll
  for (int r = 0; r < 8; ++r) {
    const int row = row0 + r + 8*half;                // global token row
    const int s   = row & (SEQ - 1);
    const int b   = row >> 11;
    const float v = acc[r];
    if (sel == 2) {
      Vt[((size_t)((b*NH + h)*HD + d)) * SEQ + s] = (__bf16)v;
    } else {
      const float p = __shfl_xor(v, 1, 32);           // partner feature (col^1)
      const int   i = d >> 1;
      const float theta = (float)s * __expf(-(float)i * 0.2878231366242557f); // ln(1e4)/32
      float sn, cs; __sincosf(theta, &sn, &cs);
      float o = (d & 1) ? (v*cs + p*sn) : (v*cs - p*sn);
      if (sel == 0) o *= 0.125f;                      // fold 1/sqrt(HD) into Q
      __bf16* dst = (sel == 0) ? Qb : Kb;
      dst[((size_t)((b*NH + h)*SEQ + s)) * HD + d] = (__bf16)o;
    }
  }
}

// ---------------- QKV GEMM (register-blocked 32x32 per wave) + RoPE ----------------
__global__ void k_qkvgemm(const __bf16* __restrict__ Xb, const __bf16* __restrict__ Wqt,
                          __bf16* __restrict__ Qb, __bf16* __restrict__ Kb,
                          __bf16* __restrict__ Vt) {
  const int lane = threadIdx.x & 31, wid = threadIdx.x >> 5;
  const int l15 = lane & 15, half = lane >> 4;
  const int NT = QKVN / 32;                           // 96
  const int tile = blockIdx.x * (blockDim.x >> 5) + wid;
  const int mt = tile / NT, nt = tile % NT;

  const __bf16* arow0 = Xb  + (size_t)(mt*32 + l15) * DM;
  const __bf16* arow1 = arow0 + (size_t)16 * DM;
  const __bf16* bcol0 = Wqt + (size_t)(nt*32 + l15) * DM;
  const __bf16* bcol1 = bcol0 + (size_t)16 * DM;

  v8f a00 = {}, a01 = {}, a10 = {}, a11 = {};
#pragma unroll 2
  for (int k0 = 0; k0 < DM; k0 += 32) {
    __builtin_prefetch(arow0 + k0 + 256, 0, 3);
    __builtin_prefetch(arow1 + k0 + 256, 0, 3);
    __builtin_prefetch(bcol0 + k0 + 256, 0, 3);
    __builtin_prefetch(bcol1 + k0 + 256, 0, 3);
    const v16bf fa0 = load_a16(arow0, k0, half);
    const v16bf fa1 = load_a16(arow1, k0, half);
    const v16bf fb0 = load_b16(bcol0, k0, half);
    const v16bf fb1 = load_b16(bcol1, k0, half);
    a00 = wmma_bf16(fa0, fb0, a00);
    a01 = wmma_bf16(fa0, fb1, a01);
    a10 = wmma_bf16(fa1, fb0, a10);
    a11 = wmma_bf16(fa1, fb1, a11);
  }

  const int m0 = mt * 32, n0 = nt * 32;
  qkv_epilogue_tile(a00, m0 +  0, n0 +  0 + l15, half, Qb, Kb, Vt);
  qkv_epilogue_tile(a01, m0 +  0, n0 + 16 + l15, half, Qb, Kb, Vt);
  qkv_epilogue_tile(a10, m0 + 16, n0 +  0 + l15, half, Qb, Kb, Vt);
  qkv_epilogue_tile(a11, m0 + 16, n0 + 16 + l15, half, Qb, Kb, Vt);
}

// ---------------- Flash-style attention ----------------
// One wave per 16-query tile of one (b,h); 32 keys per iteration.
__global__ void k_attn(const __bf16* __restrict__ Qb, const __bf16* __restrict__ Kb,
                       const __bf16* __restrict__ Vt, __bf16* __restrict__ Yb) {
  __shared__ __align__(16) __bf16 plds[8][16*32];     // per-wave P staging (C->A layout)
  const int lane = threadIdx.x & 31, wid = threadIdx.x >> 5;
  const int l15 = lane & 15, half = lane >> 4;
  const int QT = SEQ / 16;                            // 128
  const int tile = blockIdx.x * (blockDim.x >> 5) + wid;
  const int b   = tile / (NH * QT);
  const int rem = tile % (NH * QT);
  const int h   = rem / QT;
  const int q0  = (rem % QT) * 16;
  const size_t bh = (size_t)(b*NH + h);

  const __bf16* qrow = Qb + (bh*SEQ + q0 + l15) * HD;
  const v16bf qa0 = load_a16(qrow, 0,  half);
  const v16bf qa1 = load_a16(qrow, 32, half);

  __bf16* pw = &plds[wid][0];
  const __bf16* kbase = Kb + bh*SEQ*HD;
  const __bf16* vc0 = Vt + (bh*HD +  0 + l15) * SEQ;
  const __bf16* vc1 = Vt + (bh*HD + 16 + l15) * SEQ;
  const __bf16* vc2 = Vt + (bh*HD + 32 + l15) * SEQ;
  const __bf16* vc3 = Vt + (bh*HD + 48 + l15) * SEQ;

  float m_run[8], l_run[8];
  v8f o0 = {}, o1 = {}, o2 = {}, o3 = {};
#pragma unroll
  for (int r = 0; r < 8; ++r) { m_run[r] = -1e30f; l_run[r] = 0.f; }

  for (int kb = 0; kb < SEQ; kb += 32) {
    const __bf16* kcol0 = kbase + (size_t)(kb + l15) * HD;
    const __bf16* kcol1 = kcol0 + 16 * HD;
    v8f s0 = {}, s1 = {};
    s0 = wmma_bf16(qa0, load_b16(kcol0, 0,  half), s0);
    s0 = wmma_bf16(qa1, load_b16(kcol0, 32, half), s0);
    s1 = wmma_bf16(qa0, load_b16(kcol1, 0,  half), s1);
    s1 = wmma_bf16(qa1, load_b16(kcol1, 32, half), s1);

#pragma unroll
    for (int r = 0; r < 8; ++r) {
      const float a = s0[r], c = s1[r];
      float mx = fmaxf(a, c);
#pragma unroll
      for (int off = 1; off < 16; off <<= 1) mx = fmaxf(mx, __shfl_xor(mx, off, 32));
      const float mnew  = fmaxf(m_run[r], mx);
      const float scale = __expf(m_run[r] - mnew);
      const float p0 = __expf(a - mnew);
      const float p1 = __expf(c - mnew);
      float rs = p0 + p1;
#pragma unroll
      for (int off = 1; off < 16; off <<= 1) rs += __shfl_xor(rs, off, 32);
      l_run[r] = l_run[r] * scale + rs;
      m_run[r] = mnew;
      o0[r] *= scale; o1[r] *= scale; o2[r] *= scale; o3[r] *= scale;
      const int rowm = r + 8*half;                    // C-layout row owned by lane
      pw[rowm*32 + l15]      = (__bf16)p0;
      pw[rowm*32 + 16 + l15] = (__bf16)p1;
    }
    asm volatile("s_wait_dscnt 0" ::: "memory");       // P visible to whole wave
    const v16bf pa = load_a16(pw + l15*32, 0, half);   // P in A layout (16x32)
    o0 = wmma_bf16(pa, load_b16(vc0, kb, half), o0);
    o1 = wmma_bf16(pa, load_b16(vc1, kb, half), o1);
    o2 = wmma_bf16(pa, load_b16(vc2, kb, half), o2);
    o3 = wmma_bf16(pa, load_b16(vc3, kb, half), o3);
  }

#pragma unroll
  for (int r = 0; r < 8; ++r) {
    const float inv = 1.0f / l_run[r];
    const int rowm = r + 8*half;
    __bf16* y = Yb + ((size_t)(b*SEQ + q0 + rowm)) * DM + h*HD;
    y[ 0 + l15] = (__bf16)(o0[r] * inv);
    y[16 + l15] = (__bf16)(o1[r] * inv);
    y[32 + l15] = (__bf16)(o2[r] * inv);
    y[48 + l15] = (__bf16)(o3[r] * inv);
  }
}

// ---------------- Output projection (register-blocked 32x32 per wave) ----------------
__global__ void k_outproj(const __bf16* __restrict__ Yb, const __bf16* __restrict__ Wot,
                          float* __restrict__ out) {
  const int lane = threadIdx.x & 31, wid = threadIdx.x >> 5;
  const int l15 = lane & 15, half = lane >> 4;
  const int NT = DM / 32;                              // 32
  const int tile = blockIdx.x * (blockDim.x >> 5) + wid;
  const int mt = tile / NT, nt = tile % NT;

  const __bf16* arow0 = Yb  + (size_t)(mt*32 + l15) * DM;
  const __bf16* arow1 = arow0 + (size_t)16 * DM;
  const __bf16* bcol0 = Wot + (size_t)(nt*32 + l15) * DM;
  const __bf16* bcol1 = bcol0 + (size_t)16 * DM;

  v8f a00 = {}, a01 = {}, a10 = {}, a11 = {};
#pragma unroll 2
  for (int k0 = 0; k0 < DM; k0 += 32) {
    __builtin_prefetch(arow0 + k0 + 256, 0, 3);
    __builtin_prefetch(arow1 + k0 + 256, 0, 3);
    __builtin_prefetch(bcol0 + k0 + 256, 0, 3);
    __builtin_prefetch(bcol1 + k0 + 256, 0, 3);
    const v16bf fa0 = load_a16(arow0, k0, half);
    const v16bf fa1 = load_a16(arow1, k0, half);
    const v16bf fb0 = load_b16(bcol0, k0, half);
    const v16bf fb1 = load_b16(bcol1, k0, half);
    a00 = wmma_bf16(fa0, fb0, a00);
    a01 = wmma_bf16(fa0, fb1, a01);
    a10 = wmma_bf16(fa1, fb0, a10);
    a11 = wmma_bf16(fa1, fb1, a11);
  }

  const int m0 = mt * 32, n0 = nt * 32;
#pragma unroll
  for (int r = 0; r < 8; ++r) {
    const int rm = r + 8*half;
    out[(size_t)(m0 +  0 + rm) * DM + n0 +  0 + l15] = a00[r];
    out[(size_t)(m0 +  0 + rm) * DM + n0 + 16 + l15] = a01[r];
    out[(size_t)(m0 + 16 + rm) * DM + n0 +  0 + l15] = a10[r];
    out[(size_t)(m0 + 16 + rm) * DM + n0 + 16 + l15] = a11[r];
  }
}

// ---------------- host launcher ----------------
extern "C" void kernel_launch(void* const* d_in, const int* in_sizes, int n_in,
                              void* d_out, int out_size, void* d_ws, size_t ws_size,
                              hipStream_t stream) {
  (void)in_sizes; (void)n_in; (void)out_size; (void)ws_size;
  const float* x     = (const float*)d_in[0];
  const float* w_qkv = (const float*)d_in[1];
  const float* w_out = (const float*)d_in[2];
  float* out = (float*)d_out;

  char* p = (char*)d_ws;
  __bf16* Xb  = (__bf16*)p; p += (size_t)TOKENS * DM * 2;  // x in bf16
  __bf16* Wqt = (__bf16*)p; p += (size_t)QKVN  * DM  * 2;  // w_qkv^T bf16 [N][K]
  __bf16* Wot = (__bf16*)p; p += (size_t)DM    * DM  * 2;  // w_out^T bf16 [N][K]
  __bf16* Qb  = (__bf16*)p; p += (size_t)TOKENS * DM * 2;  // [b][h][s][d] roped, scaled
  __bf16* Kb  = (__bf16*)p; p += (size_t)TOKENS * DM * 2;  // [b][h][s][d] roped
  __bf16* Vt  = (__bf16*)p; p += (size_t)TOKENS * DM * 2;  // [b][h][d][s]
  __bf16* Yb  = (__bf16*)p; p += (size_t)TOKENS * DM * 2;  // attention out, bf16

  k_cvt_bf16     <<<(TOKENS*DM)/256, 256, 0, stream>>>(x, Xb, TOKENS*DM);
  k_transpose_cvt<<<(DM*QKVN)/256,  256, 0, stream>>>(w_qkv, Wqt, DM, QKVN);
  k_transpose_cvt<<<(DM*DM)/256,    256, 0, stream>>>(w_out, Wot, DM, DM);

  k_qkvgemm <<<(TOKENS/32)*(QKVN/32)/8, 256, 0, stream>>>(Xb, Wqt, Qb, Kb, Vt);
  k_attn    <<<(BATCH*NH*(SEQ/16))/8,   256, 0, stream>>>(Qb, Kb, Vt, Yb);
  k_outproj <<<(TOKENS/32)*(DM/32)/8,   256, 0, stream>>>(Yb, Wot, out);
}
